// SWAttention_15728170238371
// MI455X (gfx1250) — matrix operand order repaired and checked by reading, not compiled, not run on hardware
//
#include <hip/hip_runtime.h>
#include <hip/hip_bf16.h>

// ---------------------------------------------------------------------------
// Sliding-window attention, CDNA5 (gfx1250), wave32, WMMA bf16 + async LDS DMA.
//   B=2, S=2048, HIDDEN=2048, H=16, D=128, WINDOW=256
// Workspace layout (needs 112 MB):
//   [  0, 16M)  XB  bf16 [B*S][HID]        x converted to bf16
//   [ 16, 24M)  WQT bf16 [N][K] transposed
//   [ 24, 32M)  WKT bf16 [N][K]
//   [ 32, 40M)  WVT bf16 [N][K]
//   [ 40, 48M)  WOT bf16 [N][K]
//   [ 48, 64M)  Q   bf16 [B][H][S][128]    RoPE'd, scaled by 1/sqrt(D)*kw[h]
//   [ 64, 80M)  K   bf16 [B][H][S][128]    RoPE'd
//   [ 80, 96M)  VT  bf16 [B][H][128][S]    V transposed (dim-major)
//   [ 96,112M)  AO  bf16 [B*S][H*128]      attention output, GEMM-ready
// ---------------------------------------------------------------------------

typedef __bf16 bf16_t;
typedef __attribute__((ext_vector_type(8)))  __bf16 v8bf;
typedef __attribute__((ext_vector_type(16))) __bf16 v16bf;
typedef __attribute__((ext_vector_type(8)))  float  v8f;
typedef __attribute__((ext_vector_type(4)))  int    v4i;

#define SEQ_     2048
#define HID_     2048
#define DM_      2048
#define NH_      16
#define HD_      128
#define WIN_     256

#define AS1_ __attribute__((address_space(1)))
#define AS3_ __attribute__((address_space(3)))

#if defined(__gfx1250__) && __has_builtin(__builtin_amdgcn_global_load_async_to_lds_b128)
#define ASYNC_LDS 1
#else
#define ASYNC_LDS 0
#endif

__device__ __forceinline__ bf16_t f2bf(float f) {
  unsigned u = __builtin_bit_cast(unsigned, f);
  u += 0x7FFFu + ((u >> 16) & 1u);               // round-to-nearest-even
  unsigned short h = (unsigned short)(u >> 16);
  return __builtin_bit_cast(bf16_t, h);
}

__device__ __forceinline__ v16bf ld16(const bf16_t* lo, const bf16_t* hi) {
  v8bf a = *(const v8bf*)lo;
  v8bf b = *(const v8bf*)hi;
  return __builtin_shufflevector(a, b, 0,1,2,3,4,5,6,7,8,9,10,11,12,13,14,15);
}

__device__ __forceinline__ v8f wmma_bf16(v16bf a, v16bf b, v8f c) {
  return __builtin_amdgcn_wmma_f32_16x16x32_bf16(false, a, false, b,
                                                 (short)0, c, false, false);
}

// 16-byte global -> LDS copy: async DMA (ASYNCcnt) when available.
__device__ __forceinline__ void cp128(void* lds, const void* g) {
#if ASYNC_LDS
  __builtin_amdgcn_global_load_async_to_lds_b128(
      (AS1_ v4i*)(void*)g, (AS3_ v4i*)lds, 0, 0);
#else
  *(v8bf*)lds = *(const v8bf*)g;
#endif
}

__device__ __forceinline__ void wait_async_all() {
#if ASYNC_LDS
#if __has_builtin(__builtin_amdgcn_s_wait_asynccnt)
  __builtin_amdgcn_s_wait_asynccnt(0);
#else
  asm volatile("s_wait_asynccnt 0x0" ::: "memory");
#endif
#endif
}

// ---------------------------------------------------------------------------
// Kernel 0a: convert x to bf16. grid = 4096 x 256, 8 elem/thread.
// ---------------------------------------------------------------------------
__global__ __launch_bounds__(256) void cvt_x_kernel(
    const float* __restrict__ x, bf16_t* __restrict__ XB)
{
  size_t base = ((size_t)blockIdx.x * 256 + threadIdx.x) * 8;
  float4 f0 = *(const float4*)&x[base];
  float4 f1 = *(const float4*)&x[base + 4];
  v8bf o;
  o[0] = f2bf(f0.x); o[1] = f2bf(f0.y); o[2] = f2bf(f0.z); o[3] = f2bf(f0.w);
  o[4] = f2bf(f1.x); o[5] = f2bf(f1.y); o[6] = f2bf(f1.z); o[7] = f2bf(f1.w);
  *(v8bf*)&XB[base] = o;
}

// ---------------------------------------------------------------------------
// Kernel 0b: convert + transpose weights to bf16 [N][K]. grid = (32,32,4).
// ---------------------------------------------------------------------------
__global__ __launch_bounds__(256) void cvt_w_kernel(
    const float* __restrict__ W0, const float* __restrict__ W1,
    const float* __restrict__ W2, const float* __restrict__ W3,
    bf16_t* __restrict__ T0, bf16_t* __restrict__ T1,
    bf16_t* __restrict__ T2, bf16_t* __restrict__ T3)
{
  __shared__ __align__(16) bf16_t t[64 * 72];
  const int z = blockIdx.z;
  const float* W = (z == 0) ? W0 : (z == 1) ? W1 : (z == 2) ? W2 : W3;
  bf16_t*      T = (z == 0) ? T0 : (z == 1) ? T1 : (z == 2) ? T2 : T3;
  const int tid = threadIdx.x;
  const int k0 = blockIdx.x * 64, n0 = blockIdx.y * 64;
#pragma unroll
  for (int i = 0; i < 16; ++i) {                 // read W[k][n] coalesced
    int idx = tid + i * 256;
    int r = idx >> 6, c = idx & 63;
    t[c * 72 + r] = f2bf(W[(size_t)(k0 + r) * DM_ + n0 + c]);
  }
  __syncthreads();
#pragma unroll
  for (int i = 0; i < 2; ++i) {                  // write T[n][k] vectorized
    int vi = tid + i * 256;
    int rn = vi >> 3, c8 = (vi & 7) * 8;
    *(v8bf*)&T[(size_t)(n0 + rn) * HID_ + k0 + c8] = *(const v8bf*)&t[rn * 72 + c8];
  }
}

// ---------------------------------------------------------------------------
// Kernel 1: QKV projection GEMM (bf16) + bias + RoPE epilogue.
// grid = (32, 48); blockIdx.y/16 selects Q/K/V; 256 threads = 8 waves.
// Double-buffered LDS fed by async global->LDS DMA.
// ---------------------------------------------------------------------------
__global__ __launch_bounds__(256) void qkv_rope_gemm(
    const bf16_t* __restrict__ XB,
    const bf16_t* __restrict__ WQT, const bf16_t* __restrict__ WKT,
    const bf16_t* __restrict__ WVT,
    const float* __restrict__ bQ, const float* __restrict__ bK,
    const float* __restrict__ bV, const float* __restrict__ kw,
    bf16_t* __restrict__ Qws, bf16_t* __restrict__ Kws, bf16_t* __restrict__ VT)
{
  __shared__ __align__(16) char smem[40960];     // 2 x (ldsA 10240 + ldsB 10240)
  float* ldsC = (float*)smem;                    // [64][132], reused after GEMM

  const int tid  = threadIdx.x;
  const int lane = tid & 31;
  const int w    = tid >> 5;
  const int l15  = lane & 15;
  const int half = lane >> 4;
  const int mOff = (w & 3) * 32;
  const int nOff = (w >> 2) * 64;

  const int m0    = blockIdx.x * 128;
  const int which = blockIdx.y >> 4;             // 0=Q 1=K 2=V
  const int nIdx  = blockIdx.y & 15;             // head index
  const int n0    = nIdx * 128;

  const bf16_t* WT   = (which == 0) ? WQT : (which == 1) ? WKT : WVT;
  const float*  bias = (which == 0) ? bQ  : (which == 1) ? bK  : bV;

  const int arow = tid >> 1, acol = (tid & 1) * 16;   // 2+2 b128 per thread

  v8f acc[2][4] = {};

  auto tile_issue = [&](int buf, int k0) {
    bf16_t* ldsA = (bf16_t*)(smem + buf * 20480);
    bf16_t* ldsB = (bf16_t*)(smem + buf * 20480 + 10240);
#pragma unroll
    for (int i = 0; i < 2; ++i)
      cp128(&ldsA[arow * 40 + acol + i * 8],
            &XB[(size_t)(m0 + arow) * HID_ + k0 + acol + i * 8]);
#pragma unroll
    for (int i = 0; i < 2; ++i)
      cp128(&ldsB[arow * 40 + acol + i * 8],
            &WT[(size_t)(n0 + arow) * HID_ + k0 + acol + i * 8]);
  };

  tile_issue(0, 0);
  wait_async_all();
  __syncthreads();

  int buf = 0;
  for (int kt = 0; kt < 64; ++kt) {
    const bool more = (kt + 1) < 64;
    if (more) tile_issue(buf ^ 1, (kt + 1) * 32);   // DMA overlaps WMMAs below

    const bf16_t* ldsA = (const bf16_t*)(smem + buf * 20480);
    const bf16_t* ldsB = (const bf16_t*)(smem + buf * 20480 + 10240);
    v16bf a[2], b[4];
#pragma unroll
    for (int mt = 0; mt < 2; ++mt) {
      const bf16_t* p = &ldsA[(mOff + mt * 16 + l15) * 40 + half * 8];
      a[mt] = ld16(p, p + 16);
    }
#pragma unroll
    for (int nt = 0; nt < 4; ++nt) {
      const bf16_t* p = &ldsB[(nOff + nt * 16 + l15) * 40 + half * 8];
      b[nt] = ld16(p, p + 16);
    }
#pragma unroll
    for (int mt = 0; mt < 2; ++mt)
#pragma unroll
      for (int nt = 0; nt < 4; ++nt)
        acc[mt][nt] = wmma_bf16(a[mt], b[nt], acc[mt][nt]);

    if (more) wait_async_all();
    __syncthreads();
    buf ^= 1;
  }

  const float qscale = (which == 0) ? 0.08838834764831845f * kw[nIdx] : 1.0f;

#pragma unroll
  for (int chunk = 0; chunk < 2; ++chunk) {      // stage 64 rows at a time
    if (((w & 3) >> 1) == chunk) {
#pragma unroll
      for (int mt = 0; mt < 2; ++mt)
#pragma unroll
        for (int nt = 0; nt < 4; ++nt)
#pragma unroll
          for (int r = 0; r < 8; ++r) {
            int rl = mOff + mt * 16 + r + 8 * half - chunk * 64;
            ldsC[rl * 132 + nOff + nt * 16 + l15] = acc[mt][nt][r];
          }
    }
    __syncthreads();
    if (which < 2) {
      bf16_t* dst = (which == 0) ? Qws : Kws;
#pragma unroll
      for (int i = 0; i < 16; ++i) {             // RoPE pairs (c, c+64)
        int pidx = tid + i * 256;
        int rl = pidx >> 6;
        int c  = pidx & 63;
        int grow = m0 + chunk * 64 + rl;
        int bb = grow >> 11;
        int s  = grow & (SEQ_ - 1);
        float t1 = ldsC[rl * 132 + c]      + bias[n0 + c];
        float t2 = ldsC[rl * 132 + c + 64] + bias[n0 + c + 64];
        float ang = (float)s * __expf(-0.14391156831212787f * (float)c);
        float sn, cs;
        __sincosf(ang, &sn, &cs);
        float o1 = (t1 * cs - t2 * sn) * qscale;
        float o2 = (t1 * sn + t2 * cs) * qscale;
        size_t base = ((size_t)(bb * NH_ + nIdx) * SEQ_ + s) * HD_;
        dst[base + c]      = f2bf(o1);
        dst[base + c + 64] = f2bf(o2);
      }
    } else {
      // V: bias only, store transposed [b][h][d][s] (coalesced in s)
#pragma unroll
      for (int i = 0; i < 32; ++i) {
        int idx = tid + i * 256;                 // 64 rows x 128 dims
        int rl = idx & 63;
        int d  = idx >> 6;
        int grow = m0 + chunk * 64 + rl;
        int bb = grow >> 11;
        int s  = grow & (SEQ_ - 1);
        float v = ldsC[rl * 132 + d] + bias[n0 + d];
        VT[((size_t)(bb * NH_ + nIdx) * HD_ + d) * SEQ_ + s] = f2bf(v);
      }
    }
    __syncthreads();
  }
}

// ---------------------------------------------------------------------------
// Kernel 2: sliding-window flash attention, double-buffered K/V tiles.
// grid = (S/128, B*H); 256 threads = 8 waves, 16 queries per wave.
// ---------------------------------------------------------------------------
__global__ __launch_bounds__(256) void attn_kernel(
    const bf16_t* __restrict__ Qws, const bf16_t* __restrict__ Kws,
    const bf16_t* __restrict__ VT, bf16_t* __restrict__ AO)
{
  __shared__ __align__(16) bf16_t ldsK [2][64 * 136];   // [key][dim]
  __shared__ __align__(16) bf16_t ldsVt[2][128 * 72];   // [dim][key]
  __shared__ __align__(16) bf16_t ldsP [8 * 16 * 72];   // per-wave P tile

  const int tid  = threadIdx.x;
  const int lane = tid & 31;
  const int w    = tid >> 5;
  const int l15  = lane & 15;
  const int half = lane >> 4;

  const int q0 = blockIdx.x * 128;
  const int bh = blockIdx.y;
  const int bb = bh >> 4;
  const int h  = bh & 15;
  const bf16_t* Qp  = Qws + (size_t)bh * SEQ_ * HD_;
  const bf16_t* Kp  = Kws + (size_t)bh * SEQ_ * HD_;
  const bf16_t* VTp = VT  + (size_t)bh * HD_ * SEQ_;

  const int qw = q0 + w * 16;

  v16bf qa[4];
  const bf16_t* qrow = Qp + (size_t)(qw + l15) * HD_;
#pragma unroll
  for (int ks = 0; ks < 4; ++ks) {
    const bf16_t* p = qrow + ks * 32 + half * 8;
    qa[ks] = ld16(p, p + 16);
  }

  v8f o[8] = {};
  float mr[8], lr[8];
#pragma unroll
  for (int r = 0; r < 8; ++r) { mr[r] = -1e30f; lr[r] = 0.f; }

  int jstart = q0 - WIN_; if (jstart < 0) jstart = 0;
  jstart &= ~63;
  const int jend = q0 + 127;

  auto kv_issue = [&](int buf, int j0) {
#pragma unroll
    for (int i = 0; i < 4; ++i) {                // K tile: 64x128
      int vi = tid + i * 256;
      int key = vi >> 4, d8 = (vi & 15) * 8;
      cp128(&ldsK[buf][key * 136 + d8], &Kp[(size_t)(j0 + key) * HD_ + d8]);
    }
#pragma unroll
    for (int i = 0; i < 4; ++i) {                // V tile from VT (dim-major)
      int vi = tid + i * 256;
      int d = vi >> 3, k8 = (vi & 7) * 8;
      cp128(&ldsVt[buf][d * 72 + k8], &VTp[(size_t)d * SEQ_ + j0 + k8]);
    }
  };

  kv_issue(0, jstart);
  int kb = 0;
  for (int j0 = jstart; j0 <= jend; j0 += 64) {
    wait_async_all();
    __syncthreads();                             // tile kb ready for all waves
    if (j0 + 64 <= jend) kv_issue(kb ^ 1, j0 + 64);  // prefetch next tile

    v8f s[4] = {};
#pragma unroll
    for (int ks = 0; ks < 4; ++ks)               // S = Q . K^T  (16 WMMAs)
#pragma unroll
      for (int nt = 0; nt < 4; ++nt) {
        const bf16_t* p = &ldsK[kb][(nt * 16 + l15) * 136 + ks * 32 + half * 8];
        v16bf bk = ld16(p, p + 16);
        s[nt] = wmma_bf16(qa[ks], bk, s[nt]);
      }

    float rmax[8];
#pragma unroll
    for (int r = 0; r < 8; ++r) rmax[r] = -1e30f;
#pragma unroll
    for (int nt = 0; nt < 4; ++nt)
#pragma unroll
      for (int r = 0; r < 8; ++r) {
        int j  = j0 + nt * 16 + l15;
        int iq = qw + r + 8 * half;
        float v = s[nt][r];
        bool ok = (j <= iq) && (j >= iq - WIN_);
        v = ok ? v : -3e38f;
        s[nt][r] = v;
        rmax[r] = fmaxf(rmax[r], v);
      }
#pragma unroll
    for (int m = 1; m < 16; m <<= 1)
#pragma unroll
      for (int r = 0; r < 8; ++r)
        rmax[r] = fmaxf(rmax[r], __shfl_xor(rmax[r], m, 32));

    float fac[8], rsum[8];
#pragma unroll
    for (int r = 0; r < 8; ++r) {
      float mn = fmaxf(mr[r], rmax[r]);
      fac[r] = __expf(mr[r] - mn);
      mr[r] = mn;
      rsum[r] = 0.f;
    }
#pragma unroll
    for (int nt = 0; nt < 4; ++nt)
#pragma unroll
      for (int r = 0; r < 8; ++r) {
        float p = __expf(s[nt][r] - mr[r]);      // masked -> underflow to 0
        s[nt][r] = p;
        rsum[r] += p;
      }
#pragma unroll
    for (int m = 1; m < 16; m <<= 1)
#pragma unroll
      for (int r = 0; r < 8; ++r)
        rsum[r] += __shfl_xor(rsum[r], m, 32);
#pragma unroll
    for (int r = 0; r < 8; ++r) lr[r] = lr[r] * fac[r] + rsum[r];
#pragma unroll
    for (int ov = 0; ov < 8; ++ov)
#pragma unroll
      for (int r = 0; r < 8; ++r) o[ov][r] *= fac[r];

    bf16_t* pw = &ldsP[w * 16 * 72];             // D-layout -> A-layout via LDS
#pragma unroll
    for (int nt = 0; nt < 4; ++nt)
#pragma unroll
      for (int r = 0; r < 8; ++r)
        pw[(r + 8 * half) * 72 + nt * 16 + l15] = f2bf(s[nt][r]);

#pragma unroll
    for (int ks2 = 0; ks2 < 2; ++ks2) {          // O += P . V  (16 WMMAs)
      const bf16_t* pp = pw + l15 * 72 + ks2 * 32 + half * 8;
      v16bf pa = ld16(pp, pp + 16);
#pragma unroll
      for (int ov = 0; ov < 8; ++ov) {
        const bf16_t* vb = &ldsVt[kb][(ov * 16 + l15) * 72 + ks2 * 32 + half * 8];
        v16bf bv = ld16(vb, vb + 16);
        o[ov] = wmma_bf16(pa, bv, o[ov]);
      }
    }
    kb ^= 1;
  }

  float invl[8];
#pragma unroll
  for (int r = 0; r < 8; ++r) invl[r] = 1.0f / lr[r];
#pragma unroll
  for (int ov = 0; ov < 8; ++ov)
#pragma unroll
    for (int r = 0; r < 8; ++r) {
      int iq  = qw + r + 8 * half;
      int col = h * HD_ + ov * 16 + l15;
      AO[((size_t)bb * SEQ_ + iq) * DM_ + col] = f2bf(o[ov][r] * invl[r]);
    }
}

// ---------------------------------------------------------------------------
// Kernel 3: output projection, fp32 epilogue to d_out. grid = (32, 16).
// ---------------------------------------------------------------------------
__global__ __launch_bounds__(256) void out_proj_gemm(
    const bf16_t* __restrict__ AO, const bf16_t* __restrict__ WOT,
    const float* __restrict__ bO, float* __restrict__ out)
{
  __shared__ __align__(16) char smem[40960];

  const int tid  = threadIdx.x;
  const int lane = tid & 31;
  const int w    = tid >> 5;
  const int l15  = lane & 15;
  const int half = lane >> 4;
  const int mOff = (w & 3) * 32;
  const int nOff = (w >> 2) * 64;

  const int m0 = blockIdx.x * 128;
  const int n0 = blockIdx.y * 128;

  const int arow = tid >> 1, acol = (tid & 1) * 16;

  v8f acc[2][4] = {};

  auto tile_issue = [&](int buf, int k0) {
    bf16_t* ldsA = (bf16_t*)(smem + buf * 20480);
    bf16_t* ldsB = (bf16_t*)(smem + buf * 20480 + 10240);
#pragma unroll
    for (int i = 0; i < 2; ++i)
      cp128(&ldsA[arow * 40 + acol + i * 8],
            &AO[(size_t)(m0 + arow) * DM_ + k0 + acol + i * 8]);
#pragma unroll
    for (int i = 0; i < 2; ++i)
      cp128(&ldsB[arow * 40 + acol + i * 8],
            &WOT[(size_t)(n0 + arow) * DM_ + k0 + acol + i * 8]);
  };

  tile_issue(0, 0);
  wait_async_all();
  __syncthreads();

  int buf = 0;
  for (int kt = 0; kt < 64; ++kt) {
    const bool more = (kt + 1) < 64;
    if (more) tile_issue(buf ^ 1, (kt + 1) * 32);

    const bf16_t* ldsA = (const bf16_t*)(smem + buf * 20480);
    const bf16_t* ldsB = (const bf16_t*)(smem + buf * 20480 + 10240);
    v16bf a[2], b[4];
#pragma unroll
    for (int mt = 0; mt < 2; ++mt) {
      const bf16_t* p = &ldsA[(mOff + mt * 16 + l15) * 40 + half * 8];
      a[mt] = ld16(p, p + 16);
    }
#pragma unroll
    for (int nt = 0; nt < 4; ++nt) {
      const bf16_t* p = &ldsB[(nOff + nt * 16 + l15) * 40 + half * 8];
      b[nt] = ld16(p, p + 16);
    }
#pragma unroll
    for (int mt = 0; mt < 2; ++mt)
#pragma unroll
      for (int nt = 0; nt < 4; ++nt)
        acc[mt][nt] = wmma_bf16(a[mt], b[nt], acc[mt][nt]);

    if (more) wait_async_all();
    __syncthreads();
    buf ^= 1;
  }

#pragma unroll
  for (int mt = 0; mt < 2; ++mt)
#pragma unroll
    for (int nt = 0; nt < 4; ++nt)
#pragma unroll
      for (int r = 0; r < 8; ++r) {
        int row = m0 + mOff + mt * 16 + r + 8 * half;
        int col = n0 + nOff + nt * 16 + l15;
        out[(size_t)row * HID_ + col] = acc[mt][nt][r] + bO[col];
      }
}

// ---------------------------------------------------------------------------
extern "C" void kernel_launch(void* const* d_in, const int* in_sizes, int n_in,
                              void* d_out, int out_size, void* d_ws, size_t ws_size,
                              hipStream_t stream) {
  (void)in_sizes; (void)n_in; (void)out_size; (void)ws_size;

  const float* x  = (const float*)d_in[0];
  const float* WQ = (const float*)d_in[1];
  const float* bQ = (const float*)d_in[2];
  const float* WK = (const float*)d_in[3];
  const float* bK = (const float*)d_in[4];
  const float* WV = (const float*)d_in[5];
  const float* bV = (const float*)d_in[6];
  const float* WO = (const float*)d_in[7];
  const float* bO = (const float*)d_in[8];
  const float* kw = (const float*)d_in[9];
  float* out = (float*)d_out;

  char* ws = (char*)d_ws;
  const size_t M = 1024 * 1024;
  bf16_t* XB  = (bf16_t*)(ws);
  bf16_t* WQT = (bf16_t*)(ws + 16 * M);
  bf16_t* WKT = (bf16_t*)(ws + 24 * M);
  bf16_t* WVT = (bf16_t*)(ws + 32 * M);
  bf16_t* WOT = (bf16_t*)(ws + 40 * M);
  bf16_t* Qws = (bf16_t*)(ws + 48 * M);
  bf16_t* Kws = (bf16_t*)(ws + 64 * M);
  bf16_t* VT  = (bf16_t*)(ws + 80 * M);
  bf16_t* AO  = (bf16_t*)(ws + 96 * M);

  cvt_x_kernel <<<4096, 256, 0, stream>>>(x, XB);
  cvt_w_kernel <<<dim3(32, 32, 4), 256, 0, stream>>>(WQ, WK, WV, WO,
                                                     WQT, WKT, WVT, WOT);
  qkv_rope_gemm<<<dim3(32, 48), 256, 0, stream>>>(XB, WQT, WKT, WVT,
                                                  bQ, bK, bV, kw, Qws, Kws, VT);
  attn_kernel  <<<dim3(16, 32), 256, 0, stream>>>(Qws, Kws, VT, AO);
  out_proj_gemm<<<dim3(32, 16), 256, 0, stream>>>(AO, WOT, bO, out);
}
